// ArmNet_23871428231804
// MI455X (gfx1250) — matrix operand order repaired and checked by reading, not compiled, not run on hardware
//
#include <hip/hip_runtime.h>
#include <hip/hip_bf16.h>

typedef __attribute__((ext_vector_type(16))) _Float16 v16h;
typedef __attribute__((ext_vector_type(8)))  _Float16 v8h;
typedef __attribute__((ext_vector_type(8)))  float    v8f;

// ---------------- problem sizes (fixed by the reference) ----------------
constexpr int FRAMES = 256 * 256;       // 65536
constexpr int N1c = FRAMES * 6;         // 393216 encoder nodes
constexpr int E1c = FRAMES * 5;         // 327680 encoder edges
constexpr int N2c = FRAMES * 14;        // 917504 decoder nodes
constexpr int E2c = FRAMES * 13;        // 851968 decoder edges

// ---------------- workspace layout (bytes), aliased by liveness --------
constexpr size_t O_WT1H = 0;                       // Wt1 f16: 64*128*2
constexpr size_t O_WTRH = 16384;                   // W_tr f16: 896*384*2 = 688128
constexpr size_t O_BASE = 704512;
constexpr size_t O_H1   = O_BASE;                  // N1*16*4 = 25165824
constexpr size_t O_H2   = O_H1 + 25165824ull;      // N1*32*4 = 50331648
constexpr size_t O_H3   = O_H2 + 50331648ull;      // N1*64*4 = 100663296
constexpr size_t O_TSUM = O_H3 + 100663296ull;     // N1*64*4
constexpr size_t O_TCNT = O_TSUM + 100663296ull;   // N1*4 = 1572864
constexpr size_t O_ZH   = O_TCNT + 1572864ull;     // N1*64*2 = 50331648
constexpr size_t O_H3H  = O_BASE;                  // reuse h1+h2 (dead)
constexpr size_t O_X2   = O_BASE;                  // reuse encoder region: N2*66*4
constexpr size_t O_G1   = O_X2 + 242221056ull;     // N2*32*4 = 117440512
constexpr size_t O_G2   = O_BASE;                  // reuse x2 (dead): N2*16*4
constexpr size_t O_G3   = O_G2 + 58720256ull;      // N2*4
constexpr size_t O_DTS  = O_G3 + 3670016ull;       // N2*4
constexpr size_t O_DTC  = O_DTS + 3670016ull;      // N2*4

// ---------------- small utility kernels ----------------
__global__ void f32_to_f16_kernel(const float* __restrict__ a, _Float16* __restrict__ b, int n) {
    int i = blockIdx.x * blockDim.x + threadIdx.x;
    if (i < n) b[i] = (_Float16)a[i];
}

__global__ void count_edges_kernel(const int* __restrict__ ei, int E, float* __restrict__ cnt) {
    int e = blockIdx.x * blockDim.x + threadIdx.x;
    if (e < E) atomicAdd(&cnt[ei[E + e]], 1.0f);
}

// out = tsum/max(cnt,1) + h3, stored as f16 row-major [N1,64] == [65536,384]
__global__ void enc_temporal_fin_kernel(const float* __restrict__ tsum,
                                        const float* __restrict__ tcnt,
                                        const float* __restrict__ h3,
                                        _Float16* __restrict__ zh, int n) {
    int i = blockIdx.x * blockDim.x + threadIdx.x;
    if (i >= n) return;
    int node = i >> 6;
    float z = tsum[i] / fmaxf(tcnt[node], 1.0f) + h3[i];
    zh[i] = (_Float16)z;
}

__global__ void x2_bounds_kernel(const float* __restrict__ lower, const float* __restrict__ upper,
                                 float* __restrict__ x2, int n2) {
    int i = blockIdx.x * blockDim.x + threadIdx.x;
    if (i < n2) { x2[(size_t)i * 66 + 64] = lower[i]; x2[(size_t)i * 66 + 65] = upper[i]; }
}

// ---------------- spatial block: node residual (out = x@Wu.T + bu) -----
template <int CIN, int COUT>
__global__ void node_up_kernel(const float* __restrict__ x, const float* __restrict__ Wu,
                               const float* __restrict__ bu, float* __restrict__ out, int N) {
    int idx = blockIdx.x * blockDim.x + threadIdx.x;
    if (idx >= N * COUT) return;
    int n = idx / COUT, o = idx % COUT;
    const float* xr = x + (size_t)n * CIN;
    const float* wr = Wu + o * CIN;
    float acc = bu[o];
#pragma unroll
    for (int k = 0; k < CIN; ++k) acc = fmaf(xr[k], wr[k], acc);
    out[idx] = acc;
}

// ---------------- spatial block: edge messages + atomic segment-sum ----
template <int CIN, int EA, int COUT>
__global__ void spatial_edge_kernel(const float* __restrict__ x, const int* __restrict__ ei, int E,
                                    const float* __restrict__ eattr, const float* __restrict__ Wl,
                                    const float* __restrict__ bl, float* __restrict__ out) {
    constexpr int K = 2 * CIN + EA;
    __shared__ float Wt[K * COUT];   // transposed: Wt[k*COUT+o] = Wl[o*K+k]
    __shared__ float bs[COUT];
    for (int idx = threadIdx.x; idx < K * COUT; idx += blockDim.x) {
        int k = idx / COUT, o = idx % COUT;
        Wt[idx] = Wl[o * K + k];
    }
    for (int idx = threadIdx.x; idx < COUT; idx += blockDim.x) bs[idx] = bl[idx];
    __syncthreads();
    int e = blockIdx.x * blockDim.x + threadIdx.x;
    if (e >= E) return;
    int src = ei[e], dst = ei[E + e];
    const float* xd = x + (size_t)dst * CIN;
    const float* xs = x + (size_t)src * CIN;
    const float* ea = eattr + (size_t)e * EA;
    float acc[COUT];
#pragma unroll
    for (int o = 0; o < COUT; ++o) acc[o] = bs[o];
    for (int k = 0; k < K; ++k) {
        float zk = (k < CIN) ? xd[k] : ((k < 2 * CIN) ? xs[k - CIN] : ea[k - 2 * CIN]);
        const float* wrow = &Wt[k * COUT];
#pragma unroll
        for (int o = 0; o < COUT; ++o) acc[o] = fmaf(zk, wrow[o], acc[o]);
    }
    float* od = out + (size_t)dst * COUT;
#pragma unroll
    for (int o = 0; o < COUT; ++o) {
        float v = acc[o];
        v = (v > 0.0f) ? v : 0.01f * v;          // leaky_relu(0.01)
        atomicAdd(&od[o], v);
    }
}

// ---- async global->LDS staging helper (gfx1250 ASYNCcnt path) ----
__device__ __forceinline__ void async_copy_b128(unsigned ldsAddr, const void* gaddr) {
    unsigned long long ga = (unsigned long long)(uintptr_t)gaddr;
    asm volatile("global_load_async_to_lds_b128 %0, %1, off"
                 :: "v"(ldsAddr), "v"(ga) : "memory");
}
__device__ __forceinline__ void async_wait0() {
    asm volatile("s_wait_asynccnt 0x0" ::: "memory");
}

// ---------------- encoder temporal layer: WMMA over gathered edge rows -
// m = relu([h[dst], h[src]] @ Wt1.T + bt1) (64 outputs); tsum[dst] += m.
// 16 edges per wave tile; 4 N-tiles (64 outputs) reuse the hoisted A fragments.
__global__ void __launch_bounds__(256) temporal_wmma_kernel(
        const _Float16* __restrict__ xh,   // [N1,64] f16
        const int* __restrict__ tei,       // [2,N1]
        const _Float16* __restrict__ Bw,   // Wt1 f16 [64,128] row-major
        const float* __restrict__ bias,    // bt1 [64]
        float* __restrict__ tsum) {        // [N1,64]
    // stage Wt1 (16KB) into LDS, row stride 272B (256B data + 16B pad -> bank rotation)
    __shared__ alignas(16) _Float16 Bsh[64 * 136];
    {
        unsigned ldsBase = (unsigned)(uintptr_t)(void*)&Bsh[0];
        const char* gb = (const char*)Bw;
#pragma unroll
        for (int i = 0; i < 4; ++i) {
            unsigned c = threadIdx.x + i * 256;        // 1024 chunks of 16B
            unsigned row = c >> 4, col = c & 15;
            async_copy_b128(ldsBase + row * 272u + col * 16u, gb + row * 256u + col * 16u);
        }
        async_wait0();
    }
    __syncthreads();

    int lane = threadIdx.x & 31;
    int wave = threadIdx.x >> 5;
    int tile = blockIdx.x * 8 + wave;      // N1/16 tiles total
    int e = tile * 16 + (lane & 15);
    int dstN = tei[N1c + e];
    int srcN = tei[e];
    const _Float16* rowD = xh + (size_t)dstN * 64;
    const _Float16* rowS = xh + (size_t)srcN * 64;
    int kaOff = (lane < 16) ? 0 : 8;
    int kbOff = (lane < 16) ? 0 : 8;

    // hoist A fragments (K=128 -> 4 chunks), reused by all 4 N-tiles
    v16h afrag[4];
#pragma unroll
    for (int kc = 0; kc < 4; ++kc) {
        int k0 = kc * 32;
        const _Float16* base = (k0 < 64) ? (rowD + k0) : (rowS + (k0 - 64));
        v8h a0 = *(const v8h*)(base + kaOff);
        v8h a1 = *(const v8h*)(base + kaOff + 16);
#pragma unroll
        for (int i = 0; i < 8; ++i) { afrag[kc][i] = a0[i]; afrag[kc][i + 8] = a1[i]; }
    }
    int rbase = (lane < 16) ? 0 : 8;
    int dIdx[8];
#pragma unroll
    for (int i = 0; i < 8; ++i) dIdx[i] = tei[N1c + tile * 16 + rbase + i];

#pragma unroll
    for (int tn = 0; tn < 4; ++tn) {
        const _Float16* Bcol = &Bsh[(unsigned)(tn * 16 + (lane & 15)) * 136u];
        v8f acc = {0.f, 0.f, 0.f, 0.f, 0.f, 0.f, 0.f, 0.f};
#pragma unroll
        for (int kc = 0; kc < 4; ++kc) {
            v8h b0 = *(const v8h*)(Bcol + kc * 32 + kbOff);
            v8h b1 = *(const v8h*)(Bcol + kc * 32 + kbOff + 16);
            v16h b;
#pragma unroll
            for (int i = 0; i < 8; ++i) { b[i] = b0[i]; b[i + 8] = b1[i]; }
            acc = __builtin_amdgcn_wmma_f32_16x16x32_f16(false, afrag[kc], false, b, (short)0,
                                                         acc, false, false);
        }
        int n = tn * 16 + (lane & 15);
        float bn = bias[n];
#pragma unroll
        for (int i = 0; i < 8; ++i) {
            float v = fmaxf(acc[i] + bn, 0.0f);
            atomicAdd(&tsum[(size_t)dIdx[i] * 64 + n], v);
        }
    }
}

// ---------------- transform: [65536,384]f16 @ W_tr.T -> tanh -> x2[:, :64]
// Block = one N-tile (16 cols of W_tr staged in LDS) x 8 M-tiles (1/wave).
__global__ void __launch_bounds__(256) transform_wmma_kernel(
        const _Float16* __restrict__ A,    // zh [65536,384]
        const _Float16* __restrict__ Bw,   // W_tr f16 [896,384] row-major
        const float* __restrict__ btr,     // [896]
        float* __restrict__ x2) {          // [N2,66], cols 0..63 written here
    __shared__ alignas(16) _Float16 Bsh[16 * 392];   // 16 cols x (384+8) halves, stride 784B
    int tn = blockIdx.x >> 9;              // 56 groups of 512 blocks
    int tmBase = (blockIdx.x & 511) * 8;
    {
        unsigned ldsBase = (unsigned)(uintptr_t)(void*)&Bsh[0];
        const char* gb = (const char*)(Bw + (size_t)tn * 16 * 384);
#pragma unroll
        for (int i = 0; i < 3; ++i) {
            unsigned c = threadIdx.x + i * 256;        // 768 chunks of 16B
            unsigned row = c / 48, col = c % 48;       // 48 chunks per 768B row
            async_copy_b128(ldsBase + row * 784u + col * 16u, gb + row * 768u + col * 16u);
        }
        async_wait0();
    }
    __syncthreads();

    int lane = threadIdx.x & 31;
    int wave = threadIdx.x >> 5;
    int tm = tmBase + wave;
    const _Float16* Arow = A + (size_t)(tm * 16 + (lane & 15)) * 384;
    int kaOff = (lane < 16) ? 0 : 8;
    int kbOff = (lane < 16) ? 0 : 8;
    const _Float16* Bcol = &Bsh[(unsigned)(lane & 15) * 392u];
    v8f acc = {0.f, 0.f, 0.f, 0.f, 0.f, 0.f, 0.f, 0.f};
#pragma unroll
    for (int k0 = 0; k0 < 384; k0 += 32) {
        __builtin_prefetch(Arow + k0 + 64, 0, 1);      // global_prefetch_b8
        v8h a0 = *(const v8h*)(Arow + k0 + kaOff);
        v8h a1 = *(const v8h*)(Arow + k0 + kaOff + 16);
        v8h b0 = *(const v8h*)(Bcol + k0 + kbOff);
        v8h b1 = *(const v8h*)(Bcol + k0 + kbOff + 16);
        v16h a, b;
#pragma unroll
        for (int i = 0; i < 8; ++i) { a[i] = a0[i]; a[i + 8] = a1[i]; b[i] = b0[i]; b[i + 8] = b1[i]; }
        acc = __builtin_amdgcn_wmma_f32_16x16x32_f16(false, a, false, b, (short)0, acc,
                                                     false, false);
    }
    int n = tn * 16 + (lane & 15);
    float bn = btr[n];
    int rbase = tm * 16 + ((lane < 16) ? 0 : 8);
#pragma unroll
    for (int i = 0; i < 8; ++i) {
        size_t flat = (size_t)(rbase + i) * 896 + n;   // reshape(-1,64) -> x2 row/col
        x2[(flat >> 6) * 66 + (flat & 63)] = tanhf(acc[i] + bn);
    }
}

// ---------------- decoder temporal + output --------------------------
__global__ void dec_temporal_edge_kernel(const int* __restrict__ tei, const float* __restrict__ g3,
                                         const float* __restrict__ Wt2, const float* __restrict__ bt2,
                                         float* __restrict__ dts, float* __restrict__ dtc) {
    int e = blockIdx.x * blockDim.x + threadIdx.x;
    if (e >= N2c) return;
    int s = tei[e], d = tei[N2c + e];
    float m = fmaxf(Wt2[0] * g3[d] + Wt2[1] * g3[s] + bt2[0], 0.0f);
    atomicAdd(&dts[d], m);
    atomicAdd(&dtc[d], 1.0f);
}

__global__ void dec_final_kernel(const float* __restrict__ dts, const float* __restrict__ dtc,
                                 const float* __restrict__ g3, const float* __restrict__ lower,
                                 const float* __restrict__ upper, float* __restrict__ ang) {
    int i = blockIdx.x * blockDim.x + threadIdx.x;
    if (i >= N2c) return;
    float out = tanhf(dts[i] / fmaxf(dtc[i], 1.0f) + g3[i]);
    ang[i] = lower[i] + (upper[i] - lower[i]) * (out + 1.0f) * 0.5f;
}

// ---------------- launcher -------------------------------------------
static inline int cdiv(int a, int b) { return (a + b - 1) / b; }

extern "C" void kernel_launch(void* const* d_in, const int* in_sizes, int n_in,
                              void* d_out, int out_size, void* d_ws, size_t ws_size,
                              hipStream_t stream) {
    (void)in_sizes; (void)n_in; (void)out_size; (void)ws_size;
    const float* x     = (const float*)d_in[0];
    const int*   ei1   = (const int*)d_in[1];
    const float* ea1   = (const float*)d_in[2];
    const int*   tei1  = (const int*)d_in[3];
    const float* lower = (const float*)d_in[4];
    const float* upper = (const float*)d_in[5];
    const int*   ei2   = (const int*)d_in[6];
    const float* ea2   = (const float*)d_in[7];
    const int*   tei2  = (const int*)d_in[8];
    const float* We1_l = (const float*)d_in[9],  *be1_l = (const float*)d_in[10];
    const float* We1_u = (const float*)d_in[11], *be1_u = (const float*)d_in[12];
    const float* We2_l = (const float*)d_in[13], *be2_l = (const float*)d_in[14];
    const float* We2_u = (const float*)d_in[15], *be2_u = (const float*)d_in[16];
    const float* We3_l = (const float*)d_in[17], *be3_l = (const float*)d_in[18];
    const float* We3_u = (const float*)d_in[19], *be3_u = (const float*)d_in[20];
    const float* Wt1   = (const float*)d_in[21], *bt1   = (const float*)d_in[22];
    const float* W_tr  = (const float*)d_in[23], *b_tr  = (const float*)d_in[24];
    const float* Wd1_l = (const float*)d_in[25], *bd1_l = (const float*)d_in[26];
    const float* Wd1_u = (const float*)d_in[27], *bd1_u = (const float*)d_in[28];
    const float* Wd2_l = (const float*)d_in[29], *bd2_l = (const float*)d_in[30];
    const float* Wd2_u = (const float*)d_in[31], *bd2_u = (const float*)d_in[32];
    const float* Wd3_l = (const float*)d_in[33], *bd3_l = (const float*)d_in[34];
    const float* Wd3_u = (const float*)d_in[35], *bd3_u = (const float*)d_in[36];
    const float* Wt2   = (const float*)d_in[37], *bt2   = (const float*)d_in[38];

    char* ws = (char*)d_ws;
    _Float16* wt1h = (_Float16*)(ws + O_WT1H);
    _Float16* wtrh = (_Float16*)(ws + O_WTRH);
    float*    h1   = (float*)(ws + O_H1);
    float*    h2   = (float*)(ws + O_H2);
    float*    h3   = (float*)(ws + O_H3);
    float*    tsum = (float*)(ws + O_TSUM);
    float*    tcnt = (float*)(ws + O_TCNT);
    _Float16* zh   = (_Float16*)(ws + O_ZH);
    _Float16* h3h  = (_Float16*)(ws + O_H3H);
    float*    x2   = (float*)(ws + O_X2);
    float*    g1   = (float*)(ws + O_G1);
    float*    g2   = (float*)(ws + O_G2);
    float*    g3   = (float*)(ws + O_G3);
    float*    dts  = (float*)(ws + O_DTS);
    float*    dtc  = (float*)(ws + O_DTC);
    float*    ang  = (float*)d_out;

    const int T = 256;

    // weights -> f16 (GEMM B operands; B[k][n] = W[n][k] is column-contiguous)
    f32_to_f16_kernel<<<cdiv(64 * 128, T), T, 0, stream>>>(Wt1, wt1h, 64 * 128);
    f32_to_f16_kernel<<<cdiv(896 * 384, T), T, 0, stream>>>(W_tr, wtrh, 896 * 384);

    // encoder spatial 1..3
    node_up_kernel<6, 16><<<cdiv(N1c * 16, T), T, 0, stream>>>(x, We1_u, be1_u, h1, N1c);
    spatial_edge_kernel<6, 3, 16><<<cdiv(E1c, T), T, 0, stream>>>(x, ei1, E1c, ea1, We1_l, be1_l, h1);
    node_up_kernel<16, 32><<<cdiv(N1c * 32, T), T, 0, stream>>>(h1, We2_u, be2_u, h2, N1c);
    spatial_edge_kernel<16, 3, 32><<<cdiv(E1c, T), T, 0, stream>>>(h1, ei1, E1c, ea1, We2_l, be2_l, h2);
    node_up_kernel<32, 64><<<cdiv(N1c * 64, T), T, 0, stream>>>(h2, We3_u, be3_u, h3, N1c);
    spatial_edge_kernel<32, 3, 64><<<cdiv(E1c, T), T, 0, stream>>>(h2, ei1, E1c, ea1, We3_l, be3_l, h3);

    // encoder temporal (WMMA over gathered rows, Wt1 staged via async->LDS)
    f32_to_f16_kernel<<<cdiv(N1c * 64, T), T, 0, stream>>>(h3, h3h, N1c * 64);
    hipMemsetAsync(tsum, 0, (size_t)N1c * 64 * sizeof(float), stream);
    hipMemsetAsync(tcnt, 0, (size_t)N1c * sizeof(float), stream);
    count_edges_kernel<<<cdiv(N1c, T), T, 0, stream>>>(tei1, N1c, tcnt);
    temporal_wmma_kernel<<<(N1c / 16) / 8, T, 0, stream>>>(h3h, tei1, wt1h, bt1, tsum);
    enc_temporal_fin_kernel<<<cdiv(N1c * 64, T), T, 0, stream>>>(tsum, tcnt, h3, zh, N1c * 64);

    // transform GEMM (WMMA, B panel staged via async->LDS) -> tanh -> x2[:, :64]
    transform_wmma_kernel<<<56 * 512, T, 0, stream>>>(zh, wtrh, b_tr, x2);
    x2_bounds_kernel<<<cdiv(N2c, T), T, 0, stream>>>(lower, upper, x2, N2c);

    // decoder spatial 1..3
    node_up_kernel<66, 32><<<cdiv(N2c * 32, T), T, 0, stream>>>(x2, Wd1_u, bd1_u, g1, N2c);
    spatial_edge_kernel<66, 6, 32><<<cdiv(E2c, T), T, 0, stream>>>(x2, ei2, E2c, ea2, Wd1_l, bd1_l, g1);
    node_up_kernel<32, 16><<<cdiv(N2c * 16, T), T, 0, stream>>>(g1, Wd2_u, bd2_u, g2, N2c);
    spatial_edge_kernel<32, 6, 16><<<cdiv(E2c, T), T, 0, stream>>>(g1, ei2, E2c, ea2, Wd2_l, bd2_l, g2);
    node_up_kernel<16, 1><<<cdiv(N2c * 1, T), T, 0, stream>>>(g2, Wd3_u, bd3_u, g3, N2c);
    spatial_edge_kernel<16, 6, 1><<<cdiv(E2c, T), T, 0, stream>>>(g2, ei2, E2c, ea2, Wd3_l, bd3_l, g3);

    // decoder temporal + output mapping
    hipMemsetAsync(dts, 0, (size_t)N2c * sizeof(float), stream);
    hipMemsetAsync(dtc, 0, (size_t)N2c * sizeof(float), stream);
    dec_temporal_edge_kernel<<<cdiv(N2c, T), T, 0, stream>>>(tei2, g3, Wt2, bt2, dts, dtc);
    dec_final_kernel<<<cdiv(N2c, T), T, 0, stream>>>(dts, dtc, g3, lower, upper, ang);
}
